// LearnedConstitutiveModel_78967268704763
// MI455X (gfx1250) — matrix operand order repaired and checked by reading, not compiled, not run on hardware
//
#include <hip/hip_runtime.h>

typedef __attribute__((ext_vector_type(16))) _Float16     v16h;
typedef __attribute__((ext_vector_type(8)))  float        v8f;
typedef __attribute__((ext_vector_type(8)))  unsigned int v8u;

#define TILE          256
#define NTHREADS      256
#define ACT_STRIDE_U  68                      // 64 k-pair uints + 4 pad (16B-aligned rows)
#define ACT_STRIDE_H  136
#define ACT_U_COUNT   (TILE * ACT_STRIDE_U)   // 17408 uints per buffer
#define W2_U_COUNT    8192                    // 8 k-octets * 128 cols * 8 uints
#define SMEM_BYTES    ((2 * ACT_U_COUNT + W2_U_COUNT) * 4)  // 172032 bytes

// ---------------------------------------------------------------------------
// 2x2 SVD (closed form). Returns descending non-negative singular values and
// R = U @ Vh with the reflection (det<0) sign folded into Vh.
// ---------------------------------------------------------------------------
__device__ __forceinline__ void svd2(float a, float b, float c, float d,
                                     float& s1, float& s2,
                                     float& r00, float& r01,
                                     float& r10, float& r11) {
  float E  = 0.5f * (a + d);
  float Fq = 0.5f * (a - d);
  float G  = 0.5f * (c + b);
  float H  = 0.5f * (c - b);
  float Q  = sqrtf(E * E + H * H);
  float Rr = sqrtf(Fq * Fq + G * G);
  s1 = Q + Rr;
  float s2s = Q - Rr;
  s2 = fabsf(s2s);
  float a1 = atan2f(G, Fq);
  float a2 = atan2f(H, E);
  float th = 0.5f * (a2 - a1);   // V angle
  float ph = 0.5f * (a2 + a1);   // U angle
  float cp = cosf(ph), sp = sinf(ph);
  float ct = cosf(th), st = sinf(th);
  float g  = (s2s < 0.0f) ? -1.0f : 1.0f;
  // U = [[cp,-sp],[sp,cp]], Vh_eff = [[ct, st],[-g*st, g*ct]]
  r00 = cp * ct + g * sp * st;
  r01 = cp * st - g * sp * ct;
  r10 = sp * ct - g * cp * st;
  r11 = sp * st + g * cp * ct;
}

// ---------------------------------------------------------------------------
// Pack a 128x128 f32 weight (row-major, (out,in)) into LDS as f16 pairs,
// fragment-contiguous for the WMMA B operand:
//   wb[oct][n][j] , oct = k-pair-octet (p>>3), j = p&7, p = k>>1
//   uint = half2(W[n][2p], W[n][2p+1])
// A lane's B fragment (ks,g,col n) = 8 consecutive uints -> 2x ds_load_b128.
// ---------------------------------------------------------------------------
__device__ __forceinline__ void pack_w2(const float* __restrict__ w2,
                                        unsigned int* __restrict__ wb, int tid) {
  for (int q = tid; q < 8192; q += NTHREADS) {           // q = n*64 + p
    int n = q >> 6;
    int p = q & 63;
    float2 v = ((const float2*)w2)[q];                   // W[n][2p], W[n][2p+1]
    _Float16 h0 = (_Float16)v.x;
    _Float16 h1 = (_Float16)v.y;
    unsigned int u = (unsigned int)__builtin_bit_cast(unsigned short, h0) |
                     ((unsigned int)__builtin_bit_cast(unsigned short, h1) << 16);
    wb[((p >> 3) << 10) + (n << 3) + (p & 7)] = u;
  }
}

// ---------------------------------------------------------------------------
// Small input layer (IND -> 128) in f32, relu, write f16 activations row.
// ---------------------------------------------------------------------------
template <int IND>
__device__ __forceinline__ void layer1(const float* __restrict__ in,
                                       const float* __restrict__ w,
                                       const float* __restrict__ bias,
                                       _Float16* __restrict__ hrow) {
#pragma unroll 4
  for (int j = 0; j < 128; ++j) {
    float acc = bias[j];
#pragma unroll
    for (int i = 0; i < IND; ++i) acc += in[i] * w[j * IND + i];
    hrow[j] = (_Float16)fmaxf(acc, 0.0f);
  }
}

// ---------------------------------------------------------------------------
// Output layer (128 -> 4) in f32, no relu.
// ---------------------------------------------------------------------------
__device__ __forceinline__ void layer3(const _Float16* __restrict__ hrow,
                                       const float* __restrict__ w,
                                       const float* __restrict__ bias,
                                       float o[4]) {
  o[0] = bias[0]; o[1] = bias[1]; o[2] = bias[2]; o[3] = bias[3];
#pragma unroll 4
  for (int j = 0; j < 128; ++j) {
    float hj = (float)hrow[j];
    o[0] += hj * w[0 * 128 + j];
    o[1] += hj * w[1 * 128 + j];
    o[2] += hj * w[2 * 128 + j];
    o[3] += hj * w[3 * 128 + j];
  }
}

// ---------------------------------------------------------------------------
// Load the 16x32 f16 A fragment for rows [row base + ln], K chunk ks.
// Two ds_load_b128 per lane (contiguous k = 16*g + s within the K chunk).
// ---------------------------------------------------------------------------
__device__ __forceinline__ v16h load_a(const unsigned int* __restrict__ actp,
                                       int row, int ks, int g) {
  const uint4* p0 = (const uint4*)(actp + row * ACT_STRIDE_U + 16 * ks + 8 * g);
  const uint4* p1 = (const uint4*)(actp + row * ACT_STRIDE_U + 16 * ks + 8 * g + 4);
  uint4 lo = *p0;
  uint4 hi = *p1;
  v8u au = {lo.x, lo.y, lo.z, lo.w, hi.x, hi.y, hi.z, hi.w};
  return __builtin_bit_cast(v16h, au);
}

// ---------------------------------------------------------------------------
// Hidden 128x128 layer for one wave (rows mbase..mbase+31, two 16-row m-tiles)
// via v_wmma_f32_16x16x32_f16. A fragments hoisted (nt-invariant, 64 VGPRs);
// B fragments are two contiguous ds_load_b128; accumulators zero-init
// (inline C=0); each n-tile stored immediately with bias+relu into the other
// activation buffer.
// ---------------------------------------------------------------------------
__device__ __forceinline__ void gemm_wave(const unsigned int* __restrict__ actp_in,
                                          _Float16* __restrict__ acth_out,
                                          const unsigned int* __restrict__ wb,
                                          const float* __restrict__ bias,
                                          int mbase, int lane) {
  const int g  = lane >> 4;    // K-group select (k = 16*g + s within each 32-k step)
  const int ln = lane & 15;    // A: row M;  B/D: col N

  // Hoist all A fragments: 2 m-tiles x 4 K-steps.
  v16h a[2][4];
#pragma unroll
  for (int ks = 0; ks < 4; ++ks) {
    a[0][ks] = load_a(actp_in, mbase + ln, ks, g);
    a[1][ks] = load_a(actp_in, mbase + 16 + ln, ks, g);
  }

#pragma unroll
  for (int nt = 0; nt < 8; ++nt) {
    v8f acc0 = {};
    v8f acc1 = {};
#pragma unroll
    for (int ks = 0; ks < 4; ++ks) {
      // B fragment: 8 contiguous uints at octet (2ks+g), col nt*16+ln.
      const uint4* pb =
          (const uint4*)(wb + ((2 * ks + g) << 10) + ((nt * 16 + ln) << 3));
      uint4 b0 = pb[0];
      uint4 b1 = pb[1];
      v8u bu = {b0.x, b0.y, b0.z, b0.w, b1.x, b1.y, b1.z, b1.w};
      v16h b = __builtin_bit_cast(v16h, bu);

      acc0 = __builtin_amdgcn_wmma_f32_16x16x32_f16(
          false, a[0][ks], false, b, (short)0, acc0, false, false);
      acc1 = __builtin_amdgcn_wmma_f32_16x16x32_f16(
          false, a[1][ks], false, b, (short)0, acc1, false, false);
    }
    // D layout: VGPR k, lane -> row M = k + 8*g, col N = ln. bias+relu, f16.
    float bv = bias[nt * 16 + ln];
#pragma unroll
    for (int k = 0; k < 8; ++k) {
      float v0 = fmaxf(acc0[k] + bv, 0.0f);
      float v1 = fmaxf(acc1[k] + bv, 0.0f);
      acth_out[(mbase + k + 8 * g) * ACT_STRIDE_H + nt * 16 + ln] = (_Float16)v0;
      acth_out[(mbase + 16 + k + 8 * g) * ACT_STRIDE_H + nt * 16 + ln] = (_Float16)v1;
    }
  }
}

// ---------------------------------------------------------------------------
// Fused pipeline kernel: 256 rows / block, 8 waves, 2 m-tiles per wave.
// ---------------------------------------------------------------------------
__global__ void __launch_bounds__(NTHREADS)
lcm_kernel(const float* __restrict__ Fin,
           const float* __restrict__ dc_w1, const float* __restrict__ dc_b1,
           const float* __restrict__ dc_w2, const float* __restrict__ dc_b2,
           const float* __restrict__ dc_w3, const float* __restrict__ dc_b3,
           const float* __restrict__ sp_w1, const float* __restrict__ sp_b1,
           const float* __restrict__ sp_w2, const float* __restrict__ sp_b2,
           const float* __restrict__ sp_w3, const float* __restrict__ sp_b3,
           float* __restrict__ out, int n) {
  extern __shared__ unsigned int smem[];
  unsigned int* buf0_u = smem;                       // act ping
  unsigned int* buf1_u = smem + ACT_U_COUNT;         // act pong
  _Float16*    buf0_h  = (_Float16*)buf0_u;
  _Float16*    buf1_h  = (_Float16*)buf1_u;
  unsigned int* wb     = smem + 2 * ACT_U_COUNT;     // packed W2 (reused)

  const int tid  = threadIdx.x;
  const int lane = tid & 31;
  const int wave = tid >> 5;
  const int row  = blockIdx.x * TILE + tid;
  const int rc   = (row < n) ? row : (n - 1);        // clamp: keep EXEC uniform

  // -- stage dc_w2; per-row: load F, SVD, invariants, dc layer1 -> buf0 ------
  pack_w2(dc_w2, wb, tid);

  float4 fv = ((const float4*)Fin)[rc];
  float a = fv.x, b = fv.y, c = fv.z, d = fv.w;
  float s1, s2, r00, r01, r10, r11;
  svd2(a, b, c, d, s1, s2, r00, r01, r10, r11);
  float inv[7];
  inv[0] = s1 - 1.0f;
  inv[1] = s2 - 1.0f;
  inv[2] = a * a + c * c - 1.0f;   // (F^T F - I) row-major
  inv[3] = a * b + c * d;
  inv[4] = a * b + c * d;
  inv[5] = b * b + d * d - 1.0f;
  inv[6] = a * d - b * c - 1.0f;
  layer1<7>(inv, dc_w1, dc_b1, buf0_h + tid * ACT_STRIDE_H);
  __syncthreads();

  // -- dc hidden layer (WMMA): buf0 -> buf1 ----------------------------------
  gemm_wave(buf0_u, buf1_h, wb, dc_b2, wave * 32, lane);
  __syncthreads();

  // -- stage sp_w2; dc layer3 (buf1); Fp; 2nd SVD; sp layer1 -> buf0 ---------
  pack_w2(sp_w2, wb, tid);

  float o[4];
  layer3(buf1_h + tid * ACT_STRIDE_H, dc_w3, dc_b3, o);
  float x00 = o[0], x11 = o[3], xs = 0.5f * (o[1] + o[2]);
  float Fp00 = r00 * x00 + r01 * xs + a;
  float Fp01 = r00 * xs + r01 * x11 + b;
  float Fp10 = r10 * x00 + r11 * xs + c;
  float Fp11 = r10 * xs + r11 * x11 + d;

  float t1, t2, q00, q01, q10, q11;
  svd2(Fp00, Fp01, Fp10, Fp11, t1, t2, q00, q01, q10, q11);
  float jinv[3];
  jinv[0] = t1 + t2 - 2.0f;
  jinv[1] = Fp00 * Fp00 + Fp01 * Fp01 + Fp10 * Fp10 + Fp11 * Fp11 - 1.0f;
  jinv[2] = Fp00 * Fp11 - Fp01 * Fp10 - 1.0f;
  layer1<3>(jinv, sp_w1, sp_b1, buf0_h + tid * ACT_STRIDE_H);
  __syncthreads();

  // -- sp hidden layer (WMMA): buf0 -> buf1 ----------------------------------
  gemm_wave(buf0_u, buf1_h, wb, sp_b2, wave * 32, lane);
  __syncthreads();

  // -- sp layer3 (buf1); P = R2 @ sym(y); cauchy = P @ Fp^T ------------------
  float y[4];
  layer3(buf1_h + tid * ACT_STRIDE_H, sp_w3, sp_b3, y);
  float ys = 0.5f * (y[1] + y[2]);
  float P00 = q00 * y[0] + q01 * ys;
  float P01 = q00 * ys + q01 * y[3];
  float P10 = q10 * y[0] + q11 * ys;
  float P11 = q10 * ys + q11 * y[3];

  if (row < n) {
    float4 r;
    r.x = P00 * Fp00 + P01 * Fp01;
    r.y = P00 * Fp10 + P01 * Fp11;
    r.z = P10 * Fp00 + P11 * Fp01;
    r.w = P10 * Fp10 + P11 * Fp11;
    ((float4*)out)[row] = r;
  }
}

extern "C" void kernel_launch(void* const* d_in, const int* in_sizes, int n_in,
                              void* d_out, int out_size, void* d_ws, size_t ws_size,
                              hipStream_t stream) {
  (void)n_in; (void)out_size; (void)d_ws; (void)ws_size;
  const float* F     = (const float*)d_in[0];
  const float* dc_w1 = (const float*)d_in[1];
  const float* dc_b1 = (const float*)d_in[2];
  const float* dc_w2 = (const float*)d_in[3];
  const float* dc_b2 = (const float*)d_in[4];
  const float* dc_w3 = (const float*)d_in[5];
  const float* dc_b3 = (const float*)d_in[6];
  const float* sp_w1 = (const float*)d_in[7];
  const float* sp_b1 = (const float*)d_in[8];
  const float* sp_w2 = (const float*)d_in[9];
  const float* sp_b2 = (const float*)d_in[10];
  const float* sp_w3 = (const float*)d_in[11];
  const float* sp_b3 = (const float*)d_in[12];

  int n = in_sizes[0] / 4;                 // number of 2x2 matrices
  int blocks = (n + TILE - 1) / TILE;

  static_assert(SMEM_BYTES == 172032, "LDS layout changed");
  (void)hipFuncSetAttribute((const void*)lcm_kernel,
                            hipFuncAttributeMaxDynamicSharedMemorySize,
                            SMEM_BYTES);

  lcm_kernel<<<blocks, NTHREADS, SMEM_BYTES, stream>>>(
      F, dc_w1, dc_b1, dc_w2, dc_b2, dc_w3, dc_b3,
      sp_w1, sp_b1, sp_w2, sp_b2, sp_w3, sp_b3,
      (float*)d_out, n);
}
